// Attn_69801808495303
// MI455X (gfx1250) — matrix operand order repaired and checked by reading, not compiled, not run on hardware
//
#include <hip/hip_runtime.h>
#include <stdint.h>

#define L_DIM   2048
#define B_DIM   64
#define H_DIM   1024
#define D_DIM   2048              // 2H (K dimension)
#define F_DIM   8
#define M_DIM   (L_DIM * B_DIM)   // 131072 rows
#define K_STEP  32
#define WG_ROWS 64
#define CHUNK_N 128
#define N_CHUNKS (H_DIM / CHUNK_N)   // 8
#define K_STEPS  (D_DIM / K_STEP)    // 64

// gfx1250 async memory->LDS path for the B stream (flip to 0 if asm rejects)
#define USE_ASYNC_B 1

typedef __attribute__((ext_vector_type(16))) __bf16 v16bf;
typedef __attribute__((ext_vector_type(8)))  float  v8f;

union FragAB { v16bf bf; uint4 u[2]; };

__device__ __forceinline__ unsigned short f2bf(float f) {
  union { float f; unsigned u; } c; c.f = f;
  unsigned u = c.u;
  return (unsigned short)((u + 0x7FFFu + ((u >> 16) & 1u)) >> 16);  // RNE
}

#if USE_ASYNC_B
__device__ __forceinline__ void async_copy_b128(unsigned lds_byte_off,
                                                const void* gsrc) {
  // VDST = per-lane LDS byte address, VADDR = 64-bit global address (GV mode)
  asm volatile("global_load_async_to_lds_b128 %0, %1, off"
               :: "v"(lds_byte_off), "v"(gsrc) : "memory");
}
__device__ __forceinline__ void wait_async0() {
  asm volatile("s_wait_asynccnt 0" ::: "memory");
}
#endif

// ---------------------------------------------------------------------------
// Kernel 1: W fp32 [H][D] -> bf16 workspace (L2-resident, halves B traffic)
// ---------------------------------------------------------------------------
__global__ void wcvt_kernel(const float* __restrict__ W,
                            unsigned short* __restrict__ wb, int n4) {
  int i = blockIdx.x * blockDim.x + threadIdx.x;
  if (i < n4) {
    float4 f = ((const float4*)W)[i];
    ushort4 o;
    o.x = f2bf(f.x); o.y = f2bf(f.y); o.z = f2bf(f.z); o.w = f2bf(f.w);
    ((ushort4*)wb)[i] = o;
  }
}

// ---------------------------------------------------------------------------
// Kernel 2: fused  scores[m] = sum_h tanh(enc[m,:]·W[h,:] + b[h]) * v[h]
// WG = 256 thr (8 waves): 64 rows x N=1024 x K=2048.
// A panel (64x2048 bf16, 256KB) resident in LDS; B async double-buffered.
// ---------------------------------------------------------------------------
__global__ void __launch_bounds__(256, 1)
attn_gemm_kernel(const float* __restrict__ enc,            // [M, D] fp32
                 const unsigned short* __restrict__ wb,    // [H, D] bf16
                 const float* __restrict__ bias,           // [H]
                 const float* __restrict__ vvec,           // [H]
                 float* __restrict__ scores)               // [M]
{
  extern __shared__ unsigned short smem[];
  unsigned short* As = smem;                                  // 64*2048 bf16
  unsigned short* Bs = smem + WG_ROWS * D_DIM;                // 2*128*32 bf16
  float* s_sc = (float*)(smem + WG_ROWS * D_DIM + 2 * CHUNK_N * K_STEP);

  const int tid    = threadIdx.x;
  const int lane   = tid & 31;
  const int w      = tid >> 5;
  const int m_sub  = w >> 1;          // 0..3 -> 16-row subtile
  const int n_half = w & 1;           // 0..1 -> 64-col half of the 128 chunk
  const int hi     = (lane >= 16);
  const int ln     = lane & 15;
  const long row0  = (long)blockIdx.x * WG_ROWS;

  // ---- stage A panel: fp32 global -> bf16 LDS (coalesced float4 loads) ----
  {
    const int total4 = WG_ROWS * D_DIM / 4;     // 32768
    for (int i = tid; i < total4; i += 256) {
      int r  = i >> 9;                          // / (D/4)
      int c4 = i & 511;
      float4 f = ((const float4*)(enc + (row0 + r) * (long)D_DIM))[c4];
      ushort4 o;
      o.x = f2bf(f.x); o.y = f2bf(f.y); o.z = f2bf(f.z); o.w = f2bf(f.w);
      *((ushort4*)(As + r * D_DIM + (c4 << 2))) = o;
    }
  }
  if (tid < WG_ROWS) s_sc[tid] = 0.0f;

  float sc[8];
#pragma unroll
  for (int j = 0; j < 8; ++j) sc[j] = 0.0f;

  const v8f zacc = {};
  v8f acc0, acc1, acc2, acc3;

  const int bcol  = tid >> 1;   // 0..127: column of C == row of W in chunk
  const int bpart = tid & 1;    // which 16-element half of the 32-K slice

  // per-thread producer addresses for the B k-step tiles
#if USE_ASYNC_B
  const unsigned bs_off0 =
      (unsigned)(uintptr_t)(const void*)Bs +
      (unsigned)(2 * (bcol * K_STEP + bpart * 16));            // buf0 dest
  const unsigned buf_stride = 2 * CHUNK_N * K_STEP;            // bytes
#endif

  for (int chunk = 0; chunk < N_CHUNKS; ++chunk) {
    acc0 = zacc; acc1 = zacc; acc2 = zacc; acc3 = zacc;
    const unsigned short* wrow =
        wb + (long)(chunk * CHUNK_N + bcol) * D_DIM + bpart * 16;

    // preload B k-step 0 into buffer 0
#if USE_ASYNC_B
    async_copy_b128(bs_off0, (const void*)wrow);
    wait_async0();
#else
    {
      uint4 b0 = *((const uint4*)wrow);
      *((uint4*)(Bs + bcol * K_STEP + bpart * 16)) = b0;
    }
#endif
    __syncthreads();

    for (int ks = 0; ks < K_STEPS; ++ks) {
      const int cur = ks & 1;
#if USE_ASYNC_B
      if (ks + 1 < K_STEPS) {   // kick next tile's async copy immediately
        async_copy_b128(bs_off0 + (unsigned)(((ks + 1) & 1) * buf_stride),
                        (const void*)(wrow + (ks + 1) * K_STEP));
      }
#else
      uint4 pre;
      if (ks + 1 < K_STEPS)
        pre = *((const uint4*)(wrow + (ks + 1) * K_STEP));
#endif
      const int k0 = ks * K_STEP;

      // A fragment (contiguous k = 16*half + s, matching B)
      FragAB fa;
      const unsigned short* Arow =
          As + (m_sub * 16 + ln) * D_DIM + k0 + (hi ? 16 : 0);
      fa.u[0] = *((const uint4*)(Arow));
      fa.u[1] = *((const uint4*)(Arow + 8));

      // all four B fragments in distinct registers -> one dscnt wait,
      // then four back-to-back WMMAs on the XDL pipe
      const unsigned short* Bbase =
          Bs + cur * (CHUNK_N * K_STEP) + (n_half * 64 + ln) * K_STEP
             + (hi ? 16 : 0);
      FragAB fb0, fb1, fb2, fb3;
      fb0.u[0] = *((const uint4*)(Bbase));
      fb0.u[1] = *((const uint4*)(Bbase + 8));
      fb1.u[0] = *((const uint4*)(Bbase + 16 * K_STEP));
      fb1.u[1] = *((const uint4*)(Bbase + 16 * K_STEP + 8));
      fb2.u[0] = *((const uint4*)(Bbase + 32 * K_STEP));
      fb2.u[1] = *((const uint4*)(Bbase + 32 * K_STEP + 8));
      fb3.u[0] = *((const uint4*)(Bbase + 48 * K_STEP));
      fb3.u[1] = *((const uint4*)(Bbase + 48 * K_STEP + 8));

      acc0 = __builtin_amdgcn_wmma_f32_16x16x32_bf16(
          false, fa.bf, false, fb0.bf, (short)0, acc0, false, false);
      acc1 = __builtin_amdgcn_wmma_f32_16x16x32_bf16(
          false, fa.bf, false, fb1.bf, (short)0, acc1, false, false);
      acc2 = __builtin_amdgcn_wmma_f32_16x16x32_bf16(
          false, fa.bf, false, fb2.bf, (short)0, acc2, false, false);
      acc3 = __builtin_amdgcn_wmma_f32_16x16x32_bf16(
          false, fa.bf, false, fb3.bf, (short)0, acc3, false, false);

#if USE_ASYNC_B
      if (ks + 1 < K_STEPS) wait_async0();   // our tile landed in LDS
#else
      if (ks + 1 < K_STEPS) {
        *((uint4*)(Bs + ((ks + 1) & 1) * (CHUNK_N * K_STEP)
                      + bcol * K_STEP + bpart * 16)) = pre;
      }
#endif
      __syncthreads();
    }

    // fold this N-chunk through tanh and the v-dot (pure register work)
#pragma unroll
    for (int nt = 0; nt < 4; ++nt) {
      const int h  = chunk * CHUNK_N + n_half * 64 + nt * 16 + ln;
      const float bb = bias[h];
      const float vv = vvec[h];
      const v8f a = (nt == 0) ? acc0 : (nt == 1) ? acc1 : (nt == 2) ? acc2 : acc3;
#pragma unroll
      for (int j = 0; j < 8; ++j)
        sc[j] += tanhf(a[j] + bb) * vv;
    }
  }

  // cross-lane / cross-wave reduction: C VGPR j <-> row j (lanes 0-15) or 8+j
#pragma unroll
  for (int j = 0; j < 8; ++j) {
    int rloc = m_sub * 16 + (hi ? 8 + j : j);
    atomicAdd(&s_sc[rloc], sc[j]);     // ds_add_f32
  }
  __syncthreads();
  if (tid < WG_ROWS) scores[row0 + tid] = s_sc[tid];
}

// ---------------------------------------------------------------------------
// Kernel 3: softmax over the BATCH axis (legacy dim=0), in place.
// ---------------------------------------------------------------------------
__global__ void softmax_b_kernel(float* __restrict__ scores) {
  int l = blockIdx.x * blockDim.x + threadIdx.x;
  if (l >= L_DIM) return;
  float* s = scores + (long)l * B_DIM;
  float m = -3.402823466e38f;
  for (int b = 0; b < B_DIM; ++b) m = fmaxf(m, s[b]);
  float sum = 0.0f;
  for (int b = 0; b < B_DIM; ++b) sum += __expf(s[b] - m);
  float inv = 1.0f / sum;
  for (int b = 0; b < B_DIM; ++b) s[b] = __expf(s[b] - m) * inv;
}

// ---------------------------------------------------------------------------
// Kernel 4: broadcast to [B, F=8, L] with fully coalesced stores.
// ---------------------------------------------------------------------------
__global__ void bcast_kernel(const float* __restrict__ p,
                             float* __restrict__ out) {
  long idx = (long)blockIdx.x * blockDim.x + threadIdx.x;
  if (idx >= (long)B_DIM * F_DIM * L_DIM) return;
  int l = (int)(idx & (L_DIM - 1));
  int b = (int)(idx >> 14);            // / (F*L) = /16384
  out[idx] = p[(long)l * B_DIM + b];
}

extern "C" void kernel_launch(void* const* d_in, const int* in_sizes, int n_in,
                              void* d_out, int out_size, void* d_ws, size_t ws_size,
                              hipStream_t stream) {
  (void)in_sizes; (void)n_in; (void)out_size; (void)ws_size;
  // d_in[0] = num_features (==8, compile-time here)
  const float* enc  = (const float*)d_in[1];
  const float* W    = (const float*)d_in[2];
  const float* bias = (const float*)d_in[3];
  const float* vv   = (const float*)d_in[4];

  unsigned short* wb = (unsigned short*)d_ws;                        // 4 MB bf16 W
  float* scores = (float*)((char*)d_ws + (size_t)H_DIM * D_DIM * 2); // 512 KB
  float* out = (float*)d_out;

  const int n4 = H_DIM * D_DIM / 4;
  wcvt_kernel<<<(n4 + 255) / 256, 256, 0, stream>>>(W, wb, n4);

  const size_t smem_bytes =
      (size_t)(WG_ROWS * D_DIM + 2 * CHUNK_N * K_STEP) * 2 + WG_ROWS * 4; // 278784
  hipFuncSetAttribute((const void*)attn_gemm_kernel,
                      hipFuncAttributeMaxDynamicSharedMemorySize,
                      (int)smem_bytes);
  attn_gemm_kernel<<<M_DIM / WG_ROWS, 256, smem_bytes, stream>>>(
      enc, wb, bias, vv, scores);

  softmax_b_kernel<<<L_DIM / 256, 256, 0, stream>>>(scores);

  const long tot = (long)B_DIM * F_DIM * L_DIM;
  bcast_kernel<<<(int)((tot + 255) / 256), 256, 0, stream>>>(scores, out);
}
